// TransformerXLModule_23072564314386
// MI455X (gfx1250) — matrix operand order repaired and checked by reading, not compiled, not run on hardware
//
#include <hip/hip_runtime.h>

typedef __bf16 v16bf __attribute__((ext_vector_type(16)));
typedef float  v8f   __attribute__((ext_vector_type(8)));
typedef unsigned int v4u __attribute__((ext_vector_type(4)));
typedef int v8i __attribute__((ext_vector_type(8)));
typedef int v4i __attribute__((ext_vector_type(4)));
typedef unsigned short u16;
typedef unsigned int   u32;
typedef unsigned long long u64;
typedef long long      i64;

// Staging path selection (highest available wins):
//   USE_TDM        : tensor_load_to_lds DMA per tile (TENSORcnt)
//   USE_ASYNC_COPY : per-lane global_load_async_to_lds_b128 (ASYNCcnt)
//   else           : plain global load + ds_store
#define USE_TDM 1
#define USE_ASYNC_COPY 1

namespace txl {

constexpr int QLEN = 512, MLEN = 512, KLEN = 1024;
constexpr int NB = 16, D = 1024, H = 16, DH = 64, DI = 4096, HD = 1024;
constexpr float SCALE = 0.125f;   // 1/sqrt(64)
constexpr float EPS = 1e-5f;
constexpr float NEG_BIG = -3.0e38f;

// ---------- helpers ----------
__device__ __forceinline__ u16 f2bf(float f) {
  u32 u = __float_as_uint(f);
  u32 r = (u + 0x7FFFu + ((u >> 16) & 1u)) >> 16;   // round-to-nearest-even
  return (u16)r;
}

struct U8 { u32 u[8]; };
__device__ __forceinline__ v16bf make_frag(const uint4& lo, const uint4& hi) {
  U8 s;
  s.u[0] = lo.x; s.u[1] = lo.y; s.u[2] = lo.z; s.u[3] = lo.w;
  s.u[4] = hi.x; s.u[5] = hi.y; s.u[6] = hi.z; s.u[7] = hi.w;
  return __builtin_bit_cast(v16bf, s);
}

#if USE_TDM
// TDM descriptor-driven DMA of one 64-row x 32-bf16 tile into packed LDS.
// D# layout per CDNA5 ISA ch.8; groups 2/3 zero (2-D tensor). Invariant parts
// (group1, LDS ping-pong offsets, row base) hoisted out of the k-loop.
struct TdmTile {
  v8i g1;
  u64 gbase;
  u32 lds0, lds1;
  __device__ __forceinline__ void init(const u16* gb, int ld,
                                       const u16* l0, const u16* l1) {
    gbase = (u64)(const void*)gb;
    lds0 = (u32)(u64)(const void*)l0;
    lds1 = (u32)(u64)(const void*)l1;
    g1[0] = 0x10000;                        // data_size = 2 bytes; no multicast
    g1[1] = (int)0xFFFF0000u;               // tensor_dim0 = 0x7FFFFFFF (lo16<<16)
    g1[2] = (int)(0x7FFFu | 0xFFFF0000u);   // dim0 hi16 | tensor_dim1 lo16
    g1[3] = (int)(0x7FFFu | (32u << 16));   // dim1 hi16 | tile_dim0 = 32
    g1[4] = 64;                             // tile_dim1 = 64, tile_dim2 = 0
    g1[5] = ld;                             // tensor_dim0_stride (elems), lo32
    g1[6] = 0;                              // stride hi16 | dim1_stride lo16
    g1[7] = 0;
  }
  __device__ __forceinline__ void issue(int k0, int sel) const {
    const u64 ga = gbase + (u64)(u32)(k0 * 2);
    v4u g0;
    g0[0] = 1u;                                            // count=1, user mode
    g0[1] = sel ? lds1 : lds0;                             // lds_addr (bytes)
    g0[2] = (u32)ga;                                       // global_addr[31:0]
    g0[3] = (u32)((ga >> 32) & 0x01FFFFFFu) | (2u << 30);  // addr[56:32]|type=2
    v4i gz4 = {0, 0, 0, 0};
    v8i gz8 = {0, 0, 0, 0, 0, 0, 0, 0};
    __builtin_amdgcn_tensor_load_to_lds(g0, g1, gz4, gz4, gz8, 0);
  }
};
#endif

template <int PENDING_TILES>
__device__ __forceinline__ void wait_stage(int wave) {
#if USE_TDM
  if (wave < 2) __builtin_amdgcn_s_wait_tensorcnt(PENDING_TILES);
#elif USE_ASYNC_COPY
  if constexpr (PENDING_TILES == 0)
    asm volatile("s_wait_asynccnt 0x0" ::: "memory");
  else
    asm volatile("s_wait_asynccnt 0x2" ::: "memory");   // 2 ops per tile in flight
#else
  (void)wave;
#endif
}

// ---------- generic bf16 GEMM core ----------
// C[M,N] = A[M,K] @ B ; A row-major bf16, B stored TRANSPOSED [N,K] bf16.
// 64x64 block tile, 128 threads (4 waves, 2x2), 32x32 per wave, BK=32,
// double-buffered LDS staging.
struct GemmGeo {
  const u16* A; int lda; i64 a_batch; int a_mod;   // A[M,K]
  const u16* B; int ldb; i64 b_batch; int b_mod;   // B^T[N,K] (ldb = K-stride)
  int K;
};

template <typename Epi>
__global__ __launch_bounds__(128) void gemm_bf16_kernel(GemmGeo g, Epi epi) {
  const int bz = blockIdx.z;
  const int az = g.a_mod ? (bz % g.a_mod) : bz;
  const int bb = g.b_mod ? (bz % g.b_mod) : bz;
  const u16* A = g.A + (i64)az * g.a_batch;
  const u16* B = g.B + (i64)bb * g.b_batch;

  const int m0 = blockIdx.y * 64;
  const int n0 = blockIdx.x * 64;

  __shared__ __align__(16) u16 As[2][64 * 32];   // [m][k] ping-pong
  __shared__ __align__(16) u16 Bs[2][64 * 32];   // [n][k] ping-pong

  const int tid  = threadIdx.x;
  const int lane = tid & 31;
  // wave index is uniform across the wave: force it into an SGPR so the
  // TDM-issue gates become scalar branches (no EXEC save/restore).
  const int wave = __builtin_amdgcn_readfirstlane(tid >> 5);
  const int wm = wave >> 1, wn = wave & 1;    // 2x2 wave grid
  const int half = lane >> 4, lm = lane & 15;

  v8f acc[2][2] = {};

  const int tr = tid >> 1, tc = (tid & 1) * 16;
  (void)tr; (void)tc;

#if USE_TDM
  TdmTile tdm;
  if (wave == 0)      tdm.init(&A[(i64)m0 * g.lda], g.lda, &As[0][0], &As[1][0]);
  else if (wave == 1) tdm.init(&B[(i64)n0 * g.ldb], g.ldb, &Bs[0][0], &Bs[1][0]);
#endif

  auto issue_tile = [&](int k0, int sel) {
#if USE_TDM
    if (wave < 2) tdm.issue(k0, sel);
#elif USE_ASYNC_COPY
    const u16* gA = &A[(i64)(m0 + tr) * g.lda + (k0 + tc)];
    const u16* gB = &B[(i64)(n0 + tr) * g.ldb + (k0 + tc)];
    const u32 ldsA = (u32)(u64)(const void*)&As[sel][tr * 32 + tc];
    const u32 ldsB = (u32)(u64)(const void*)&Bs[sel][tr * 32 + tc];
    asm volatile("global_load_async_to_lds_b128 %0, %1, off"
                 :: "v"(ldsA), "v"((u64)gA) : "memory");
    asm volatile("global_load_async_to_lds_b128 %0, %1, off"
                 :: "v"(ldsB), "v"((u64)gB) : "memory");
#else
    *(uint4*)&As[sel][tr * 32 + tc] =
        *(const uint4*)&A[(i64)(m0 + tr) * g.lda + (k0 + tc)];
    *(uint4*)&Bs[sel][tr * 32 + tc] =
        *(const uint4*)&B[(i64)(n0 + tr) * g.ldb + (k0 + tc)];
#endif
  };

  const int T = g.K / 32;
  issue_tile(0, 0);
  for (int t = 0; t < T; ++t) {
    const int sel = t & 1;
    if (t + 1 < T) {
      issue_tile((t + 1) * 32, sel ^ 1);   // prefetch next tile into other buffer
      wait_stage<1>(wave);                 // current tile complete; next in flight
    } else {
      wait_stage<0>(wave);
    }
    __syncthreads();

    v16bf af[2], bfr[2];
#pragma unroll
    for (int mi = 0; mi < 2; ++mi) {
      // A 16x32: VGPR 0-3 <- K = half*8 .. +7 ; VGPR 4-7 <- K = 16+half*8 .. +7
      const int row = wm * 32 + mi * 16 + lm;
      const uint4 lo = *(const uint4*)&As[sel][row * 32 + half * 8];
      const uint4 hi = *(const uint4*)&As[sel][row * 32 + 16 + half * 8];
      af[mi] = make_frag(lo, hi);
    }
#pragma unroll
    for (int ni = 0; ni < 2; ++ni) {
      // B 32x16: lane holds column n; VGPR 0-3 <- K = half*8 .. +7 ; VGPR 4-7 <- K = 16+half*8 .. +7
      const int col = wn * 32 + ni * 16 + lm;
      const uint4 lo = *(const uint4*)&Bs[sel][col * 32 + half * 8];
      const uint4 hi = *(const uint4*)&Bs[sel][col * 32 + 16 + half * 8];
      bfr[ni] = make_frag(lo, hi);
    }

#pragma unroll
    for (int mi = 0; mi < 2; ++mi)
#pragma unroll
      for (int ni = 0; ni < 2; ++ni)
        acc[mi][ni] = __builtin_amdgcn_wmma_f32_16x16x32_bf16(
            false, af[mi], false, bfr[ni], (short)0, acc[mi][ni], false, false);
    __syncthreads();
  }

  // epilogue: C element (m = r + 8*half, n = lane&15) per VGPR r
#pragma unroll
  for (int mi = 0; mi < 2; ++mi)
#pragma unroll
    for (int ni = 0; ni < 2; ++ni) {
      const int col = n0 + wn * 32 + ni * 16 + lm;
#pragma unroll
      for (int r = 0; r < 8; ++r) {
        const int row = m0 + wm * 32 + mi * 16 + r + half * 8;
        epi(bz, row, col, acc[mi][ni][r]);
      }
    }
}

// ---------- epilogues ----------
struct EpiQKV {  // rows: m = j*NB + b over klen; cols: [q|k|v] * (h*64+d)
  u16 *qrw, *qrr, *kb, *vt;
  const float *rwb, *rrb;
  __device__ void operator()(int, int row, int col, float acc) const {
    const int j = row >> 4, b = row & 15;       // NB == 16
    const int which = col >> 10, c = col & 1023;
    const int h = c >> 6, d = c & 63;
    if (which == 0) {
      if (j >= MLEN) {
        const int i = j - MLEN;
        const i64 o = ((((i64)b * H + h) * QLEN) + i) * DH + d;
        qrw[o] = f2bf(acc + rwb[h * DH + d]);
        qrr[o] = f2bf(acc + rrb[h * DH + d]);
      }
    } else if (which == 1) {
      kb[((((i64)b * H + h) * KLEN) + j) * DH + d] = f2bf(acc);  // K (== B^T for AC)
    } else {
      vt[((((i64)b * H + h) * DH) + d) * KLEN + j] = f2bf(acc);  // V^T (B^T for PV)
    }
  }
};

struct EpiRkt {  // r_k -> [h][l][d]  (== B^T for BD)
  u16* rkb;
  __device__ void operator()(int, int row, int col, float acc) const {
    const int h = col >> 6, d = col & 63;
    rkb[((i64)h * KLEN + row) * DH + d] = f2bf(acc);
  }
};

struct EpiAC {  // batched over z = b*H+h : S[z][i][j] = acc
  float* S;
  __device__ void operator()(int bz, int row, int col, float acc) const {
    S[((i64)bz * QLEN + row) * KLEN + col] = acc;
  }
};

struct EpiBD {  // rel_shift fused: BD[i][j] = raw[i][l], j = l - (QLEN-1) + i
  float* S;
  __device__ void operator()(int bz, int row, int col, float acc) const {
    const int j = col - (QLEN - 1) + row;
    if ((unsigned)j < (unsigned)KLEN)
      S[((i64)bz * QLEN + row) * KLEN + j] += acc;
  }
};

struct EpiPV {  // attn_vec[i*NB+b][h*64+d]
  u16* av;
  __device__ void operator()(int bz, int row, int col, float acc) const {
    const int b = bz >> 4, h = bz & 15;
    av[((i64)row * NB + b) * HD + h * DH + col] = f2bf(acc);
  }
};

struct EpiStoreF32 {
  float* C; int ldc;
  __device__ void operator()(int, int row, int col, float acc) const {
    C[(i64)row * ldc + col] = acc;
  }
};

struct EpiBiasReluBf {  // FF1: relu(acc + b1) -> bf16
  u16* C; const float* bias; int ldc;
  __device__ void operator()(int, int row, int col, float acc) const {
    float y = acc + bias[col];
    C[(i64)row * ldc + col] = f2bf(y > 0.f ? y : 0.f);
  }
};

struct EpiBiasF32 {  // FF2: acc + b2 -> fp32
  float* C; const float* bias; int ldc;
  __device__ void operator()(int, int row, int col, float acc) const {
    C[(i64)row * ldc + col] = acc + bias[col];
  }
};

// ---------- elementwise / reduction kernels ----------
__global__ __launch_bounds__(256) void cast_f32_bf16_kernel(const float* __restrict__ src,
                                                            u16* __restrict__ dst, i64 n) {
  i64 idx = (i64)blockIdx.x * blockDim.x + threadIdx.x;
  const i64 stride = (i64)gridDim.x * blockDim.x;
  for (; idx < n; idx += stride) dst[idx] = f2bf(src[idx]);
}

// dst[N,K] = bf16(src[K,N]) : tiled transpose-cast for weight matrices
__global__ __launch_bounds__(256) void transpose_cast_kernel(const float* __restrict__ src,
                                                             u16* __restrict__ dst,
                                                             int K, int N) {
  __shared__ float tile[32][33];
  const int x = threadIdx.x;            // 0..31
  const int y = threadIdx.y;            // 0..7
  const int k0 = blockIdx.y * 32, n0 = blockIdx.x * 32;
#pragma unroll
  for (int i = 0; i < 32; i += 8)
    tile[y + i][x] = src[(i64)(k0 + y + i) * N + (n0 + x)];
  __syncthreads();
#pragma unroll
  for (int i = 0; i < 32; i += 8)
    dst[(i64)(n0 + y + i) * K + (k0 + x)] = f2bf(tile[x][y + i]);
}

// one wave per score row: scale, causal-with-memory mask, softmax,
// write bf16 probs in place over the fp32 row (read-before-write, row-local)
__global__ __launch_bounds__(256) void softmax_kernel(float* __restrict__ S) {
  const int wave = threadIdx.x >> 5, lane = threadIdx.x & 31;
  const i64 row = (i64)blockIdx.x * 8 + wave;
  const int i = (int)(row & (QLEN - 1));
  float* srow = S + row * KLEN;

  float vals[32];
  float mx = NEG_BIG;
#pragma unroll
  for (int t = 0; t < 32; ++t) {
    const int j = lane + t * 32;
    float s = srow[j];
    s = (j <= i + MLEN) ? s * SCALE : NEG_BIG;
    vals[t] = s;
    mx = fmaxf(mx, s);
  }
#pragma unroll
  for (int o = 16; o > 0; o >>= 1) mx = fmaxf(mx, __shfl_xor(mx, o, 32));

  float sum = 0.f;
#pragma unroll
  for (int t = 0; t < 32; ++t) { vals[t] = __expf(vals[t] - mx); sum += vals[t]; }
#pragma unroll
  for (int o = 16; o > 0; o >>= 1) sum += __shfl_xor(sum, o, 32);
  const float inv = 1.0f / sum;

  u16* prow = (u16*)srow;
#pragma unroll
  for (int t = 0; t < 32; ++t) prow[lane + t * 32] = f2bf(vals[t] * inv);
}

// out = LayerNorm(xa + xb) ; optional fp32 and bf16 outputs
__global__ __launch_bounds__(256) void add_ln_kernel(const float* __restrict__ xa,
                                                     const float* __restrict__ xb,
                                                     const float* __restrict__ g,
                                                     const float* __restrict__ bt,
                                                     float* __restrict__ outf,
                                                     u16* __restrict__ outbf) {
  const int row = blockIdx.x;
  const float* a = xa + (i64)row * D;
  const float* b = xb + (i64)row * D;
  __shared__ float r1[256], r2[256];

  float x[4], s = 0.f, s2 = 0.f;
#pragma unroll
  for (int t = 0; t < 4; ++t) {
    const int c = threadIdx.x + t * 256;
    x[t] = a[c] + b[c];
    s += x[t];
    s2 += x[t] * x[t];
  }
  r1[threadIdx.x] = s; r2[threadIdx.x] = s2;
  __syncthreads();
  for (int st = 128; st > 0; st >>= 1) {
    if (threadIdx.x < st) { r1[threadIdx.x] += r1[threadIdx.x + st];
                            r2[threadIdx.x] += r2[threadIdx.x + st]; }
    __syncthreads();
  }
  const float mean = r1[0] * (1.0f / D);
  const float var  = r2[0] * (1.0f / D) - mean * mean;
  const float rstd = rsqrtf(var + EPS);
#pragma unroll
  for (int t = 0; t < 4; ++t) {
    const int c = threadIdx.x + t * 256;
    const float y = (x[t] - mean) * rstd * g[c] + bt[c];
    if (outf)  outf[(i64)row * D + c] = y;
    if (outbf) outbf[(i64)row * D + c] = f2bf(y);
  }
}

}  // namespace txl

// ---------- host launch ----------
extern "C" void kernel_launch(void* const* d_in, const int* in_sizes, int n_in,
                              void* d_out, int out_size, void* d_ws, size_t ws_size,
                              hipStream_t stream) {
  using namespace txl;
  (void)in_sizes; (void)n_in; (void)out_size; (void)ws_size;

  const float* content = (const float*)d_in[0];
  const float* mems    = (const float*)d_in[1];
  const float* rel_pos = (const float*)d_in[2];
  const float* W_qkv   = (const float*)d_in[3];
  const float* W_r     = (const float*)d_in[4];
  const float* W_o     = (const float*)d_in[5];
  const float* rwb     = (const float*)d_in[6];
  const float* rrb     = (const float*)d_in[7];
  const float* ln1_g   = (const float*)d_in[8];
  const float* ln1_b   = (const float*)d_in[9];
  const float* ff_W1   = (const float*)d_in[10];
  const float* ff_b1   = (const float*)d_in[11];
  const float* ff_W2   = (const float*)d_in[12];
  const float* ff_b2   = (const float*)d_in[13];
  const float* ln2_g   = (const float*)d_in[14];
  const float* ln2_b   = (const float*)d_in[15];
  float* out = (float*)d_out;

  char* wsp = (char*)d_ws;
  auto take = [&](i64 bytes) -> char* {
    char* p = wsp;
    wsp += (bytes + 255) & ~(i64)255;
    return p;
  };

  constexpr i64 MROWS = (i64)KLEN * NB;          // 16384
  constexpr i64 QROWS = (i64)QLEN * NB;          // 8192

  u16*   XCAT   = (u16*)  take(MROWS * D * 2);           // [16384,1024] bf16 (A)
  u16*   WQKVT  = (u16*)  take((i64)3 * HD * D * 2);     // [3072,1024]  (B^T)
  u16*   WRT    = (u16*)  take((i64)HD * D * 2);         // [1024,1024]  (B^T)
  u16*   RPB    = (u16*)  take((i64)KLEN * D * 2);       // [1024,1024]  (A)
  u16*   WOT    = (u16*)  take((i64)D * HD * 2);         // [1024,1024]  (B^T)
  u16*   W1T    = (u16*)  take((i64)DI * D * 2);         // [4096,1024]  (B^T)
  u16*   W2T    = (u16*)  take((i64)D * DI * 2);         // [1024,4096]  (B^T)
  u16*   QRW    = (u16*)  take((i64)NB * H * QLEN * DH * 2);
  u16*   QRR    = (u16*)  take((i64)NB * H * QLEN * DH * 2);
  u16*   KB     = (u16*)  take((i64)NB * H * KLEN * DH * 2);  // K[b][h][j][d]
  u16*   VT     = (u16*)  take((i64)NB * H * DH * KLEN * 2);  // V^T[b][h][d][j]
  u16*   RKB    = (u16*)  take((i64)H * KLEN * DH * 2);       // r_k[h][l][d]
  float* S      = (float*)take((i64)NB * H * QLEN * KLEN * 4); // 512 MB, P aliases it
  u16*   AVEC   = (u16*)  take(QROWS * HD * 2);
  float* ATTNO  = (float*)take(QROWS * D * 4);
  float* OUT1   = (float*)take(QROWS * D * 4);
  u16*   OUT1B  = (u16*)  take(QROWS * D * 2);
  u16*   H1     = (u16*)  take(QROWS * DI * 2);
  float* FF2O   = (float*)take(QROWS * D * 4);

  const dim3 CB(256);
  const dim3 CG(4096);
  const dim3 TB(32, 8);
  // bf16 casts (Xcat = [mems;content], already row-contiguous as j*B+b)
  cast_f32_bf16_kernel<<<CG, CB, 0, stream>>>(mems, XCAT, (i64)MLEN * NB * D);
  cast_f32_bf16_kernel<<<CG, CB, 0, stream>>>(content, XCAT + (i64)MLEN * NB * D,
                                              (i64)QLEN * NB * D);
  cast_f32_bf16_kernel<<<CG, CB, 0, stream>>>(rel_pos, RPB, (i64)KLEN * D);
  // transposed bf16 weights (B^T operands)
  transpose_cast_kernel<<<dim3(3 * HD / 32, D / 32), TB, 0, stream>>>(W_qkv, WQKVT, D, 3 * HD);
  transpose_cast_kernel<<<dim3(HD / 32, D / 32), TB, 0, stream>>>(W_r, WRT, D, HD);
  transpose_cast_kernel<<<dim3(D / 32, HD / 32), TB, 0, stream>>>(W_o, WOT, HD, D);
  transpose_cast_kernel<<<dim3(DI / 32, D / 32), TB, 0, stream>>>(ff_W1, W1T, D, DI);
  transpose_cast_kernel<<<dim3(D / 32, DI / 32), TB, 0, stream>>>(ff_W2, W2T, DI, D);

  // 1) QKV projection: [16384,1024] x [1024,3072] with fused bias/reshape scatter
  {
    GemmGeo g{XCAT, D, 0, 0, WQKVT, D, 0, 0, D};
    EpiQKV e{QRW, QRR, KB, VT, rwb, rrb};
    gemm_bf16_kernel<EpiQKV><<<dim3(3 * HD / 64, MROWS / 64, 1), 128, 0, stream>>>(g, e);
  }
  // 2) r_k = rel_pos @ W_r, scattered per head as B^T
  {
    GemmGeo g{RPB, D, 0, 0, WRT, D, 0, 0, D};
    EpiRkt e{RKB};
    gemm_bf16_kernel<EpiRkt><<<dim3(HD / 64, KLEN / 64, 1), 128, 0, stream>>>(g, e);
  }
  // 3) AC scores, batched over 256 (b,h): Qrw[512,64] @ K^T -> B^T = K[j][d]
  {
    GemmGeo g{QRW, DH, (i64)QLEN * DH, 0, KB, DH, (i64)KLEN * DH, 0, DH};
    EpiAC e{S};
    gemm_bf16_kernel<EpiAC><<<dim3(KLEN / 64, QLEN / 64, NB * H), 128, 0, stream>>>(g, e);
  }
  // 4) BD scores with fused rel_shift (r_k shared across batch via b_mod)
  {
    GemmGeo g{QRR, DH, (i64)QLEN * DH, 0, RKB, DH, (i64)KLEN * DH, H, DH};
    EpiBD e{S};
    gemm_bf16_kernel<EpiBD><<<dim3(KLEN / 64, QLEN / 64, NB * H), 128, 0, stream>>>(g, e);
  }
  // 5) masked softmax; bf16 probs written in place into S
  softmax_kernel<<<dim3((NB * H * QLEN) / 8), 256, 0, stream>>>(S);
  // 6) attn_vec = P @ V  (P bf16, fp32-row stride 2*KLEN; B^T = V^T[d][j])
  {
    GemmGeo g{(const u16*)S, 2 * KLEN, (i64)QLEN * 2 * KLEN, 0,
              VT, KLEN, (i64)DH * KLEN, 0, KLEN};
    EpiPV e{AVEC};
    gemm_bf16_kernel<EpiPV><<<dim3(DH / 64, QLEN / 64, NB * H), 128, 0, stream>>>(g, e);
  }
  // 7) attn_out = attn_vec @ W_o
  {
    GemmGeo g{AVEC, HD, 0, 0, WOT, HD, 0, 0, HD};
    EpiStoreF32 e{ATTNO, D};
    gemm_bf16_kernel<EpiStoreF32><<<dim3(D / 64, QROWS / 64, 1), 128, 0, stream>>>(g, e);
  }
  // 8) out1 = LN(content + attn_out)
  add_ln_kernel<<<dim3((int)QROWS), 256, 0, stream>>>(content, ATTNO, ln1_g, ln1_b,
                                                      OUT1, OUT1B);
  // 9) h1 = relu(out1 @ W1 + b1)
  {
    GemmGeo g{OUT1B, D, 0, 0, W1T, D, 0, 0, D};
    EpiBiasReluBf e{H1, ff_b1, DI};
    gemm_bf16_kernel<EpiBiasReluBf><<<dim3(DI / 64, QROWS / 64, 1), 128, 0, stream>>>(g, e);
  }
  // 10) ff = h1 @ W2 + b2
  {
    GemmGeo g{H1, DI, 0, 0, W2T, DI, 0, 0, DI};
    EpiBiasF32 e{FF2O, ff_b2, D};
    gemm_bf16_kernel<EpiBiasF32><<<dim3(D / 64, QROWS / 64, 1), 128, 0, stream>>>(g, e);
  }
  // 11) out = LN(out1 + ff)
  add_ln_kernel<<<dim3((int)QROWS), 256, 0, stream>>>(OUT1, FF2O, ln2_g, ln2_b,
                                                      out, nullptr);
}
